// HiFiMambaUnit2D_31593779429652
// MI455X (gfx1250) — matrix operand order repaired and checked by reading, not compiled, not run on hardware
//
#include <hip/hip_runtime.h>
#include <cstdint>

// =====================================================================
// HiFiMambaUnit2D for MI455X (gfx1250, wave32).
//  * All 1x1 convs -> pre-swizzled-weight WMMA GEMMs (f16 in, f32 acc).
//  * GEMM operands staged into LDS via the Tensor Data Mover
//    (tensor_load_to_lds + s_wait_tensorcnt), ds_load-fed WMMA loop.
//  * Whole-sample GroupNorm via deterministic 2-stage tree reduction.
//  * Mamba selective scan -> 3-pass chunked linear-recurrence scan
//    (288 chunks of 128 steps; parallel over 4608 blocks).
//  * Workspace requirement: ~500 MB.
// =====================================================================

typedef __attribute__((ext_vector_type(16))) _Float16 v16h;
typedef __attribute__((ext_vector_type(8)))  _Float16 v8h;
typedef __attribute__((ext_vector_type(8)))  float    v8f;
typedef __attribute__((ext_vector_type(4)))  unsigned int u32x4;
typedef __attribute__((ext_vector_type(8)))  int i32x8;
typedef __attribute__((ext_vector_type(4)))  int i32x4;

#define DEVI __device__ __forceinline__

// TDM builtin availability (device pass only; host pass falls back).
#ifdef __has_builtin
#  if __has_builtin(__builtin_amdgcn_tensor_load_to_lds) && \
      __has_builtin(__builtin_amdgcn_s_wait_tensorcnt)
#    define HAS_TDM 1
#  endif
#endif
#ifndef HAS_TDM
#  define HAS_TDM 0
#endif

namespace cfg {
constexpr int Bz     = 2;
constexpr int DIM    = 128;
constexpr int HALFC  = 64;
constexpr int Hh     = 192;
constexpr int Wd     = 192;
constexpr int HW     = Hh * Wd;        // 36864 == sequence length L per batch
constexpr int NPOS   = Bz * HW;        // 73728 positions total
constexpr int XPW    = 48;             // xproj out width padded 36 -> 48
constexpr int CL     = 128;            // scan chunk length
constexpr int NCHUNK = HW / CL;        // 288 chunks per batch
constexpr int LOWW   = 96;             // pooled spatial size
}

DEVI float gelu_f(float x)    { return 0.5f * x * (1.f + erff(x * 0.70710678118654752f)); }
DEVI float sigmoid_f(float x) { return 1.f / (1.f + __expf(-x)); }

#if HAS_TDM
// 1-D TDM copy: nqw 8-byte units from global 'ga' to LDS offset 'lds_off'.
// D# per CDNA5 ISA ch.8: group0 = {count|flags, lds_addr, global_addr, type=2},
// group1 = {data_size=8B, tensor_dim0=nqw, tensor_dim1=1, tile_dim0=nqw,
//           tile_dim1/2=0, tensor_dim0_stride=nqw}; groups 2/3 unused.
DEVI void tdm_load_1d(unsigned long long ga, unsigned lds_off, unsigned nqw) {
  u32x4 g0 = {0u, 0u, 0u, 0u};
  g0[0] = 1u;                                            // count = 1, user mode
  g0[1] = lds_off;                                       // LDS byte address
  g0[2] = (unsigned)(ga & 0xffffffffull);                // global_addr[31:0]
  g0[3] = (unsigned)((ga >> 32) & 0x1ffffffull) | (2u << 30);  // [56:32] | type=2
  i32x8 g1 = {0, 0, 0, 0, 0, 0, 0, 0};
  g1[0] = (int)(3u << 16);                               // data_size = 8 bytes
  g1[1] = (int)((nqw & 0xffffu) << 16);                  // tensor_dim0 lo16
  g1[2] = (int)(((nqw >> 16) & 0xffffu) | (1u << 16));   // td0 hi16 | tensor_dim1 = 1
  g1[3] = (int)((nqw & 0xffffu) << 16);                  // tile_dim0 (fits 16 bits)
  g1[5] = (int)nqw;                                      // tensor_dim0_stride lo32
  i32x4 z4 = {0, 0, 0, 0};
#if __clang_major__ >= 23
  i32x8 z8 = {0, 0, 0, 0, 0, 0, 0, 0};
  __builtin_amdgcn_tensor_load_to_lds(g0, g1, z4, z4, z8, 0);
#else
  __builtin_amdgcn_tensor_load_to_lds(g0, g1, z4, z4, 0);
#endif
}
#endif

// ---------------------------------------------------------------------
// Weight pre-swizzle: W[COUT,CIN] f32 (row-major) -> packed f16 in the
// exact WMMA B-operand lane layout:
//   flat = ((nt*KT + kt)*32 + lane)*16 + e ; n = nt*16+(lane&15) ;
//   k = kt*32 + (e>>3)*16 + (lane>>4)*8 + (e&7)
// (lane%16 == N column; 16 packed K values per lane in table order)
// ---------------------------------------------------------------------
__global__ __launch_bounds__(256) void pack_w(const float* __restrict__ w,
                                              _Float16* __restrict__ dst,
                                              int cout, int cin) {
  int i    = blockIdx.x * 256 + threadIdx.x;
  int e    = i & 15;
  int lane = (i >> 4) & 31;
  int KT   = cin / 32;
  int kt   = (i >> 9) % KT;
  int nt   = (i >> 9) / KT;
  int n    = nt * 16 + (lane & 15);
  int k    = kt * 32 + ((e >> 3) << 4) + ((lane >> 4) << 3) + (e & 7);
  float v  = (n < cout) ? w[(size_t)n * cin + k] : 0.f;
  dst[i]   = (_Float16)v;
}

// ---------------------------------------------------------------------
// WMMA GEMM: out[NPOS, NT*16] = A[NPOS, CIN] * Wpacked^T ; f32 acc.
// 256 threads = 8 waves; each wave owns a 16-row M stripe, all N tiles
// held as v8f accumulators in VGPRs.  Both operands staged into LDS by
// the Tensor Data Mover; inner loop is ds_load_b128 + v_wmma only.
// mode: 0 = none, 1 = exact GELU.  Optional f16 mirror of the output.
// ---------------------------------------------------------------------
template <int CIN, int NT>
__global__ __launch_bounds__(256) void gemm_wmma(const _Float16* __restrict__ A,
                                                 const _Float16* __restrict__ Wp,
                                                 float* __restrict__ out,
                                                 _Float16* __restrict__ outh,
                                                 int mode) {
  constexpr int KT    = CIN / 32;
  constexpr int OUTW  = NT * 16;
  constexpr int WELEM = NT * 16 * CIN;   // packed weight f16 elements
  constexpr int AELEM = 128 * CIN;       // A stripe f16 elements (contiguous)
  static_assert(WELEM * 2 / 8 <= 65535 && AELEM * 2 / 8 <= 65535, "tile_dim0");

  __shared__ __align__(32) _Float16 sm[WELEM + AELEM];
  _Float16* sB = sm;
  _Float16* sA = sm + WELEM;

  const _Float16* Ablk = A + (size_t)blockIdx.x * 128 * CIN;
  __builtin_prefetch(A + ((size_t)blockIdx.x + 1) * 128 * CIN, 0, 1);

#if HAS_TDM
  if (threadIdx.x < 32) {
    tdm_load_1d((unsigned long long)(uintptr_t)Wp, 0u, (unsigned)(WELEM * 2 / 8));
    tdm_load_1d((unsigned long long)(uintptr_t)Ablk, (unsigned)(WELEM * 2),
                (unsigned)(AELEM * 2 / 8));
    __builtin_amdgcn_s_wait_tensorcnt(0);
  }
#else
  for (int i = threadIdx.x; i < WELEM; i += 256) sB[i] = Wp[i];
  for (int i = threadIdx.x; i < AELEM; i += 256) sA[i] = Ablk[i];
#endif
  __syncthreads();

  const int lane   = threadIdx.x & 31;
  const int mbase  = blockIdx.x * 128 + (threadIdx.x >> 5) * 16;
  const int mlocal = (threadIdx.x >> 5) * 16 + (lane & 15);
  const int ksel   = (lane >> 4) << 3;               // 0 or 8
  const _Float16* arow = sA + mlocal * CIN + ksel;

  v8f acc[NT];
#pragma unroll
  for (int nt = 0; nt < NT; ++nt)
#pragma unroll
    for (int i = 0; i < 8; ++i) acc[nt][i] = 0.f;

#pragma unroll
  for (int kt = 0; kt < KT; ++kt) {
    v8h lo = *reinterpret_cast<const v8h*>(arow + kt * 32);
    v8h hi = *reinterpret_cast<const v8h*>(arow + kt * 32 + 16);
    v16h a = __builtin_shufflevector(lo, hi, 0, 1, 2, 3, 4, 5, 6, 7,
                                     8, 9, 10, 11, 12, 13, 14, 15);
#pragma unroll
    for (int nt = 0; nt < NT; ++nt) {
      v16h b = reinterpret_cast<const v16h*>(sB)[(nt * KT + kt) * 32 + lane];
      acc[nt] = __builtin_amdgcn_wmma_f32_16x16x32_f16(
          false, a, false, b, (short)0, acc[nt], false, false);
    }
  }

  const int rsel = (lane >> 4) << 3;
  const int col  = lane & 15;
#pragma unroll
  for (int nt = 0; nt < NT; ++nt)
#pragma unroll
    for (int r = 0; r < 8; ++r) {
      float v = acc[nt][r];
      if (mode == 1) v = gelu_f(v);
      size_t o = (size_t)(mbase + rsel + r) * OUTW + nt * 16 + col;
      out[o] = v;
      if (outh) outh[o] = (_Float16)v;
    }
}

// ---------------------------------------------------------------------
// GroupNorm (whole-sample, 1 group) -- deterministic 2-stage reduction.
// ---------------------------------------------------------------------
__global__ __launch_bounds__(256) void gn_partial(const float* __restrict__ t,
                                                  float* __restrict__ part) {
  size_t idx = (size_t)blockIdx.x * 256 + threadIdx.x;
  float v = t[idx];
  float s = v, q = v * v;
#pragma unroll
  for (int m = 16; m >= 1; m >>= 1) { s += __shfl_xor(s, m, 32); q += __shfl_xor(q, m, 32); }
  __shared__ float ss[8], sq[8];
  if ((threadIdx.x & 31) == 0) { ss[threadIdx.x >> 5] = s; sq[threadIdx.x >> 5] = q; }
  __syncthreads();
  if (threadIdx.x == 0) {
    float S = 0.f, Q = 0.f;
    for (int i = 0; i < 8; ++i) { S += ss[i]; Q += sq[i]; }
    part[(size_t)blockIdx.x * 2]     = S;
    part[(size_t)blockIdx.x * 2 + 1] = Q;
  }
}

__global__ __launch_bounds__(256) void gn_reduce(const float* __restrict__ part,
                                                 float* __restrict__ stats,
                                                 int blocksPerBatch) {
  int b = blockIdx.x;
  float S = 0.f, Q = 0.f;
  for (int i = threadIdx.x; i < blocksPerBatch; i += 256) {
    S += part[(size_t)(b * blocksPerBatch + i) * 2];
    Q += part[(size_t)(b * blocksPerBatch + i) * 2 + 1];
  }
#pragma unroll
  for (int m = 16; m >= 1; m >>= 1) { S += __shfl_xor(S, m, 32); Q += __shfl_xor(Q, m, 32); }
  __shared__ float ss[8], sq[8];
  if ((threadIdx.x & 31) == 0) { ss[threadIdx.x >> 5] = S; sq[threadIdx.x >> 5] = Q; }
  __syncthreads();
  if (threadIdx.x == 0) {
    float SS = 0.f, QQ = 0.f;
    for (int i = 0; i < 8; ++i) { SS += ss[i]; QQ += sq[i]; }
    stats[b * 2] = SS; stats[b * 2 + 1] = QQ;
  }
}

__global__ __launch_bounds__(256) void gn_apply_gelu(const float* __restrict__ t,
                                                     const float* __restrict__ stats,
                                                     const float* __restrict__ g,
                                                     const float* __restrict__ bb,
                                                     _Float16* __restrict__ outh,
                                                     int C, size_t perBatch, float invCnt) {
  size_t idx = (size_t)blockIdx.x * 256 + threadIdx.x;
  int b = (int)(idx / perBatch);
  int c = (int)(idx % C);
  float mean = stats[b * 2] * invCnt;
  float var  = stats[b * 2 + 1] * invCnt - mean * mean;
  float v = (t[idx] - mean) * rsqrtf(var + 1e-5f) * g[c] + bb[c];
  outh[idx] = (_Float16)gelu_f(v);
}

// ---------------------------------------------------------------------
// Depthwise 3x3 (zero pad), f16 activations in, f32 out. NHWC layout.
// ---------------------------------------------------------------------
template <int C>
__global__ __launch_bounds__(256) void dw3x3_k(const _Float16* __restrict__ in,
                                               const float* __restrict__ wd,
                                               float* __restrict__ out) {
  size_t idx = (size_t)blockIdx.x * 256 + threadIdx.x;
  int c = (int)(idx % C);
  size_t g = idx / C;
  int b = (int)(g / cfg::HW);
  int p = (int)(g % cfg::HW);
  int h = p / cfg::Wd, w = p % cfg::Wd;
  float acc = 0.f;
#pragma unroll
  for (int dy = -1; dy <= 1; ++dy) {
    int nh = h + dy;
    if (nh < 0 || nh >= cfg::Hh) continue;
#pragma unroll
    for (int dx = -1; dx <= 1; ++dx) {
      int nw = w + dx;
      if (nw < 0 || nw >= cfg::Wd) continue;
      size_t gi = (size_t)b * cfg::HW + nh * cfg::Wd + nw;
      acc += (float)in[gi * C + c] * wd[c * 9 + (dy + 1) * 3 + (dx + 1)];
    }
  }
  out[idx] = acc;
}

// --------------------------- elementwise -----------------------------
__global__ __launch_bounds__(256) void split_pack(const float* __restrict__ x,
                                                  _Float16* __restrict__ lo,
                                                  _Float16* __restrict__ hi) {
  size_t idx = (size_t)blockIdx.x * 256 + threadIdx.x;   // N*64
  int c = (int)(idx & 63);
  size_t g = idx >> 6;
  int b = (int)(g / cfg::HW);
  int p = (int)(g % cfg::HW);
  lo[idx] = (_Float16)x[((size_t)(b * cfg::DIM + c)) * cfg::HW + p];
  hi[idx] = (_Float16)x[((size_t)(b * cfg::DIM + cfg::HALFC + c)) * cfg::HW + p];
}

__global__ __launch_bounds__(256) void add_resx(const float* __restrict__ Gm,
                                                const float* __restrict__ x,
                                                float* __restrict__ out,
                                                _Float16* __restrict__ outh, int coff) {
  size_t idx = (size_t)blockIdx.x * 256 + threadIdx.x;   // N*64
  int c = (int)(idx & 63);
  size_t g = idx >> 6;
  int b = (int)(g / cfg::HW);
  int p = (int)(g % cfg::HW);
  float v = Gm[idx] + x[((size_t)(b * cfg::DIM + coff + c)) * cfg::HW + p];
  out[idx] = v;
  if (outh) outh[idx] = (_Float16)v;
}

__global__ __launch_bounds__(256) void add2(const float* __restrict__ a,
                                            const float* __restrict__ bsrc,
                                            float* __restrict__ out,
                                            _Float16* __restrict__ outh) {
  size_t idx = (size_t)blockIdx.x * 256 + threadIdx.x;
  float v = a[idx] + bsrc[idx];
  out[idx] = v;
  if (outh) outh[idx] = (_Float16)v;
}

__global__ __launch_bounds__(256) void pool2x2(const float* __restrict__ xw,
                                               float* __restrict__ low) {
  size_t idx = (size_t)blockIdx.x * 256 + threadIdx.x;   // B*96*96*64
  int c = (int)(idx & 63);
  size_t r = idx >> 6;
  int b = (int)(r / (cfg::LOWW * cfg::LOWW));
  int q = (int)(r % (cfg::LOWW * cfg::LOWW));
  int i = q / cfg::LOWW, j = q % cfg::LOWW;
  size_t g0 = (size_t)b * cfg::HW + (2 * i) * cfg::Wd + 2 * j;
  float v = 0.25f * (xw[g0 * 64 + c] + xw[(g0 + 1) * 64 + c] +
                     xw[(g0 + cfg::Wd) * 64 + c] + xw[(g0 + cfg::Wd + 1) * 64 + c]);
  low[idx] = v;
}

__global__ __launch_bounds__(256) void up_high(const float* __restrict__ low,
                                               const float* __restrict__ xw,
                                               float* __restrict__ lowup,
                                               float* __restrict__ high) {
  size_t idx = (size_t)blockIdx.x * 256 + threadIdx.x;   // N*64
  int c = (int)(idx & 63);
  size_t g = idx >> 6;
  int b = (int)(g / cfg::HW);
  int p = (int)(g % cfg::HW);
  int h = p / cfg::Wd, w = p % cfg::Wd;
  float sh = h * 0.5f - 0.25f, sw = w * 0.5f - 0.25f;
  float fh = floorf(sh), fw = floorf(sw);
  float ah = sh - fh, aw = sw - fw;
  int h0 = min(cfg::LOWW - 1, max(0, (int)fh));
  int h1 = min(cfg::LOWW - 1, max(0, (int)fh + 1));
  int w0 = min(cfg::LOWW - 1, max(0, (int)fw));
  int w1 = min(cfg::LOWW - 1, max(0, (int)fw + 1));
  size_t base = (size_t)b * cfg::LOWW * cfg::LOWW;
  auto L = [&](int i2, int j2) {
    return low[(base + (size_t)i2 * cfg::LOWW + j2) * 64 + c];
  };
  float v = (1.f - ah) * ((1.f - aw) * L(h0, w0) + aw * L(h0, w1)) +
            ah * ((1.f - aw) * L(h1, w0) + aw * L(h1, w1));
  lowup[idx] = v;
  high[idx] = xw[idx] - v;
}

// per-position LayerNorm over 64 channels; one wave per position
__global__ __launch_bounds__(256) void seq_ln(const float* __restrict__ lowup,
                                              const float* __restrict__ sg,
                                              const float* __restrict__ lng,
                                              const float* __restrict__ lnb,
                                              _Float16* __restrict__ outh) {
  int lane = threadIdx.x & 31;
  size_t g = (size_t)blockIdx.x * 8 + (threadIdx.x >> 5);
  size_t base = g * 64;
  float v0 = lowup[base + lane] + sg[base + lane];
  float v1 = lowup[base + lane + 32] + sg[base + lane + 32];
  float s = v0 + v1, q = v0 * v0 + v1 * v1;
#pragma unroll
  for (int m = 16; m >= 1; m >>= 1) { s += __shfl_xor(s, m, 32); q += __shfl_xor(q, m, 32); }
  float mean = s * (1.f / 64.f);
  float var  = q * (1.f / 64.f) - mean * mean;
  float r = rsqrtf(var + 1e-5f);
  outh[base + lane]      = (_Float16)((v0 - mean) * r * lng[lane]      + lnb[lane]);
  outh[base + lane + 32] = (_Float16)((v1 - mean) * r * lng[lane + 32] + lnb[lane + 32]);
}

// causal depthwise conv1d (K=4) over the flattened spatial sequence + SiLU
__global__ __launch_bounds__(256) void conv1d_silu(const float* __restrict__ xz,
                                                   const float* __restrict__ cw,
                                                   const float* __restrict__ cb,
                                                   float* __restrict__ XS,
                                                   _Float16* __restrict__ XSh) {
  size_t idx = (size_t)blockIdx.x * 256 + threadIdx.x;   // N*128
  int d = (int)(idx & 127);
  size_t g = idx >> 7;
  int b = (int)(g / cfg::HW);
  int l = (int)(g % cfg::HW);
  float acc = cb[d];
#pragma unroll
  for (int k = 0; k < 4; ++k) {
    int li = l - 3 + k;
    if (li >= 0) acc += cw[d * 4 + k] * xz[((size_t)b * cfg::HW + li) * 256 + d];
  }
  float s = acc * sigmoid_f(acc);
  XS[idx] = s;
  XSh[idx] = (_Float16)s;
}

__global__ __launch_bounds__(256) void dt_softplus(const float* __restrict__ dbl,
                                                   const float* __restrict__ dtw,
                                                   const float* __restrict__ dtb,
                                                   float* __restrict__ DT) {
  size_t idx = (size_t)blockIdx.x * 256 + threadIdx.x;   // N*128
  int d = (int)(idx & 127);
  size_t g = idx >> 7;
  float s = dtb[d];
#pragma unroll
  for (int r = 0; r < 4; ++r) s += dbl[g * cfg::XPW + r] * dtw[d * 4 + r];
  DT[idx] = (s > 20.f) ? s : log1pf(__expf(s));
}

// ------------------ chunked selective-scan (3 passes) ----------------
__global__ __launch_bounds__(256) void scan_chunk(const float* __restrict__ DT,
                                                  const float* __restrict__ XS,
                                                  const float* __restrict__ DBL,
                                                  const float* __restrict__ Alog,
                                                  float* __restrict__ P,
                                                  float* __restrict__ Q) {
  int blk = blockIdx.x;
  int dg = blk & 7; blk >>= 3;
  int ch = blk % cfg::NCHUNK;
  int b  = blk / cfg::NCHUNK;
  int d = dg * 16 + (threadIdx.x >> 4);
  int n = threadIdx.x & 15;
  float A = -__expf(Alog[d * 16 + n]);
  float p = 1.f, q = 0.f;
  size_t gbase = (size_t)b * cfg::HW + (size_t)ch * cfg::CL;
  for (int l = 0; l < cfg::CL; ++l) {
    size_t g = gbase + l;
    float dt = DT[g * 128 + d];
    float x  = XS[g * 128 + d];
    float Bv = DBL[g * cfg::XPW + 4 + n];
    float dA = __expf(dt * A);
    q = dA * q + (dA - 1.f) / A * Bv * x;
    p *= dA;
  }
  size_t o = (((size_t)b * cfg::NCHUNK + ch) * 2048) + d * 16 + n;
  P[o] = p; Q[o] = q;
}

__global__ __launch_bounds__(256) void scan_carry(const float* __restrict__ P,
                                                  const float* __restrict__ Q,
                                                  float* __restrict__ Hini) {
  int t = blockIdx.x * 256 + threadIdx.x;   // 4096 states
  int b = t >> 11;
  int dn = t & 2047;
  float h = 0.f;
  for (int c = 0; c < cfg::NCHUNK; ++c) {
    size_t o = (((size_t)b * cfg::NCHUNK + c) * 2048) + dn;
    Hini[o] = h;
    h = P[o] * h + Q[o];
  }
}

__global__ __launch_bounds__(256) void scan_emit(const float* __restrict__ DT,
                                                 const float* __restrict__ XS,
                                                 const float* __restrict__ DBL,
                                                 const float* __restrict__ Alog,
                                                 const float* __restrict__ Hini,
                                                 float* __restrict__ Y) {
  int blk = blockIdx.x;
  int dg = blk & 7; blk >>= 3;
  int ch = blk % cfg::NCHUNK;
  int b  = blk / cfg::NCHUNK;
  int d = dg * 16 + (threadIdx.x >> 4);
  int n = threadIdx.x & 15;
  float A = -__expf(Alog[d * 16 + n]);
  size_t o0 = (((size_t)b * cfg::NCHUNK + ch) * 2048) + d * 16 + n;
  float h = Hini[o0];
  size_t gbase = (size_t)b * cfg::HW + (size_t)ch * cfg::CL;
  for (int l = 0; l < cfg::CL; ++l) {
    size_t g = gbase + l;
    float dt = DT[g * 128 + d];
    float x  = XS[g * 128 + d];
    float Bv = DBL[g * cfg::XPW + 4 + n];
    float Cv = DBL[g * cfg::XPW + 20 + n];
    float dA = __expf(dt * A);
    h = dA * h + (dA - 1.f) / A * Bv * x;
    float part = h * Cv;
    part += __shfl_xor(part, 8, 16);
    part += __shfl_xor(part, 4, 16);
    part += __shfl_xor(part, 2, 16);
    part += __shfl_xor(part, 1, 16);
    if (n == 0) Y[g * 128 + d] = part;
  }
}

__global__ __launch_bounds__(256) void ssm_epilogue(const float* __restrict__ Y,
                                                    const float* __restrict__ XS,
                                                    const float* __restrict__ Dp,
                                                    const float* __restrict__ xz,
                                                    _Float16* __restrict__ Yh) {
  size_t idx = (size_t)blockIdx.x * 256 + threadIdx.x;   // N*128
  int d = (int)(idx & 127);
  size_t g = idx >> 7;
  float y = Y[idx] + XS[idx] * Dp[d];
  float z = xz[g * 256 + 128 + d];
  y *= z * sigmoid_f(z);
  Yh[idx] = (_Float16)y;
}

__global__ __launch_bounds__(256) void concat_h(const float* __restrict__ lo,
                                                const float* __restrict__ hi,
                                                _Float16* __restrict__ cat) {
  size_t idx = (size_t)blockIdx.x * 256 + threadIdx.x;   // N*128
  int c = (int)(idx & 127);
  size_t g = idx >> 7;
  float v = (c < 64) ? lo[g * 64 + c] : hi[g * 64 + (c - 64)];
  cat[idx] = (_Float16)v;
}

__global__ __launch_bounds__(256) void ca_apply(const float* __restrict__ dwout,
                                                const float* __restrict__ fused,
                                                _Float16* __restrict__ outh) {
  size_t idx = (size_t)blockIdx.x * 256 + threadIdx.x;   // N*128
  float a = sigmoid_f(dwout[idx]);
  outh[idx] = (_Float16)(a * fused[idx]);
}

__global__ __launch_bounds__(256) void final_add(const float* __restrict__ x,
                                                 const float* __restrict__ outg,
                                                 float* __restrict__ dout) {
  size_t idx = (size_t)blockIdx.x * 256 + threadIdx.x;   // N*128
  int c = (int)(idx & 127);
  size_t g = idx >> 7;
  int b = (int)(g / cfg::HW);
  int p = (int)(g % cfg::HW);
  size_t xo = ((size_t)(b * cfg::DIM + c)) * cfg::HW + p;
  dout[xo] = x[xo] + outg[idx];
}

// =====================================================================
// Host orchestration
// =====================================================================
struct WInfo { int in_idx, cout, cin, nt; };
static const WInfo wtab[13] = {
    {1, 256, 64, 16},  // wl_pre.w1
    {7, 64, 256, 4},   // wl_pre.w2
    {8, 256, 64, 16},  // crm.w1
    {14, 64, 256, 4},  // crm.w2
    {15, 128, 64, 8},  // hr.w1
    {21, 64, 128, 4},  // hr.w2
    {24, 64, 64, 4},   // guide_w
    {25, 256, 64, 16}, // in_w
    {28, 36, 128, 3},  // xproj_w (padded to 48)
    {33, 64, 128, 4},  // out_ssm_w
    {34, 128, 128, 8}, // fuse_w
    {35, 128, 128, 8}, // ca_w1
    {37, 128, 128, 8}, // out_w
};

extern "C" void kernel_launch(void* const* d_in, const int* in_sizes, int n_in,
                              void* d_out, int out_size, void* d_ws, size_t ws_size,
                              hipStream_t stream) {
  (void)in_sizes; (void)n_in; (void)out_size; (void)ws_size;
  using namespace cfg;
  const float* x = (const float*)d_in[0];
#define FIN(i) ((const float*)d_in[(i)])

  char* ws = (char*)d_ws;
  size_t off = 0;
  auto alloc = [&](size_t bytes) -> char* {
    char* p = ws + off;
    off += (bytes + 255) & ~(size_t)255;
    return p;
  };

  float* stats = (float*)alloc(16 * 4);
  float* part  = (float*)alloc((size_t)(NPOS) * 2 * 4);  // >= max GN partial blocks

  _Float16* wpk[13];
  for (int i = 0; i < 13; ++i)
    wpk[i] = (_Float16*)alloc((size_t)wtab[i].nt * 16 * wtab[i].cin * 2);

  _Float16* xlow_h  = (_Float16*)alloc((size_t)NPOS * 64 * 2);
  _Float16* xhigh_h = (_Float16*)alloc((size_t)NPOS * 64 * 2);
  float*    T       = (float*)alloc((size_t)NPOS * 256 * 4);
  _Float16* Th_a    = (_Float16*)alloc((size_t)NPOS * 256 * 2);
  _Float16* Th_b    = (_Float16*)alloc((size_t)NPOS * 256 * 2);
  float*    G64     = (float*)alloc((size_t)NPOS * 64 * 4);
  float*    xw      = (float*)alloc((size_t)NPOS * 64 * 4);
  float*    guide   = (float*)alloc((size_t)NPOS * 64 * 4);
  _Float16* guide_h = (_Float16*)alloc((size_t)NPOS * 64 * 2);
  float*    lowb    = (float*)alloc((size_t)Bz * LOWW * LOWW * 64 * 4);
  float*    lowup   = (float*)alloc((size_t)NPOS * 64 * 4);
  float*    high    = (float*)alloc((size_t)NPOS * 64 * 4);
  float*    hg      = (float*)alloc((size_t)NPOS * 64 * 4);
  _Float16* hg_h    = (_Float16*)alloc((size_t)NPOS * 64 * 2);
  _Float16* seq_h   = (_Float16*)alloc((size_t)NPOS * 64 * 2);
  float*    XS      = (float*)alloc((size_t)NPOS * 128 * 4);
  _Float16* XSh     = (_Float16*)alloc((size_t)NPOS * 128 * 2);
  float*    DT      = (float*)alloc((size_t)NPOS * 128 * 4);
  float*    DBL     = (float*)alloc((size_t)NPOS * XPW * 4);
  float*    Pb      = (float*)alloc((size_t)Bz * NCHUNK * 2048 * 4);
  float*    Qb      = (float*)alloc((size_t)Bz * NCHUNK * 2048 * 4);
  float*    Hini    = (float*)alloc((size_t)Bz * NCHUNK * 2048 * 4);
  float*    Y       = (float*)alloc((size_t)NPOS * 128 * 4);
  _Float16* Yh      = (_Float16*)alloc((size_t)NPOS * 128 * 2);

  // time-multiplexed aliases (lifetimes verified against schedule below)
  float*    low_enh  = DT;                         // after scan passes done
  float*    high_enh = high;                       // after hg is formed
  _Float16* cat_hb   = (_Float16*)Y;               // after ssm_epilogue
  float*    FUSE     = T;                          // after bneck3 done
  float*    CAg      = T + (size_t)NPOS * 128;     // second half of T
  _Float16* fuse_h   = Th_a;
  _Float16* ca_h     = Th_b;
  float*    CAdw     = XS;
  _Float16* f2_h     = XSh;
  float*    OUTG     = DT;                         // after concat consumed

  const unsigned GE  = NPOS * 64 / 256;    // elementwise grids
  const unsigned GE2 = NPOS * 128 / 256;
  const unsigned GG  = NPOS / 128;         // gemm grid (576)

  auto groupnorm = [&](const float* t, int C, const float* g, const float* b,
                       _Float16* outh) {
    unsigned blocks = (unsigned)((size_t)NPOS * C / 256);
    gn_partial<<<blocks, 256, 0, stream>>>(t, part);
    gn_reduce<<<2, 256, 0, stream>>>(part, stats, (int)(blocks / 2));
    float invCnt = 1.f / ((float)HW * (float)C);
    gn_apply_gelu<<<blocks, 256, 0, stream>>>(t, stats, g, b, outh, C,
                                              (size_t)HW * C, invCnt);
  };

  // -- weight pre-swizzle into WMMA B-operand layout --
  for (int i = 0; i < 13; ++i) {
    unsigned tot = (unsigned)(wtab[i].nt * 16 * wtab[i].cin);
    pack_w<<<tot / 256, 256, 0, stream>>>(FIN(wtab[i].in_idx), wpk[i],
                                          wtab[i].cout, wtab[i].cin);
  }

  split_pack<<<GE, 256, 0, stream>>>(x, xlow_h, xhigh_h);

  // ---- bneck wl_pre (low branch) ----
  gemm_wmma<64, 16><<<GG, 256, 0, stream>>>(xlow_h, wpk[0], T, nullptr, 0);
  groupnorm(T, 256, FIN(2), FIN(3), Th_a);
  dw3x3_k<256><<<NPOS * 256 / 256, 256, 0, stream>>>(Th_a, FIN(4), T);
  groupnorm(T, 256, FIN(5), FIN(6), Th_b);
  gemm_wmma<256, 4><<<GG, 256, 0, stream>>>(Th_b, wpk[1], G64, nullptr, 1);
  add_resx<<<GE, 256, 0, stream>>>(G64, x, xw, nullptr, 0);

  pool2x2<<<(unsigned)((size_t)Bz * LOWW * LOWW * 64 / 256), 256, 0, stream>>>(xw, lowb);
  up_high<<<GE, 256, 0, stream>>>(lowb, xw, lowup, high);

  // ---- bneck crm (guide) ----
  gemm_wmma<64, 16><<<GG, 256, 0, stream>>>(xhigh_h, wpk[2], T, nullptr, 0);
  groupnorm(T, 256, FIN(9), FIN(10), Th_a);
  dw3x3_k<256><<<NPOS * 256 / 256, 256, 0, stream>>>(Th_a, FIN(11), T);
  groupnorm(T, 256, FIN(12), FIN(13), Th_b);
  gemm_wmma<256, 4><<<GG, 256, 0, stream>>>(Th_b, wpk[3], G64, nullptr, 1);
  add_resx<<<GE, 256, 0, stream>>>(G64, x, guide, guide_h, HALFC);

  // ---- seq = LN(low_up + guide @ guide_w) ----
  gemm_wmma<64, 4><<<GG, 256, 0, stream>>>(guide_h, wpk[6], G64, nullptr, 0);
  seq_ln<<<NPOS / 8, 256, 0, stream>>>(lowup, G64, FIN(22), FIN(23), seq_h);

  // ---- fcssm ----
  gemm_wmma<64, 16><<<GG, 256, 0, stream>>>(seq_h, wpk[7], T, nullptr, 0);  // xz
  conv1d_silu<<<GE2, 256, 0, stream>>>(T, FIN(26), FIN(27), XS, XSh);
  gemm_wmma<128, 3><<<GG, 256, 0, stream>>>(XSh, wpk[8], DBL, nullptr, 0);
  dt_softplus<<<GE2, 256, 0, stream>>>(DBL, FIN(29), FIN(30), DT);
  scan_chunk<<<Bz * NCHUNK * 8, 256, 0, stream>>>(DT, XS, DBL, FIN(31), Pb, Qb);
  scan_carry<<<16, 256, 0, stream>>>(Pb, Qb, Hini);
  scan_emit<<<Bz * NCHUNK * 8, 256, 0, stream>>>(DT, XS, DBL, FIN(31), Hini, Y);
  ssm_epilogue<<<GE2, 256, 0, stream>>>(Y, XS, FIN(32), T, Yh);
  gemm_wmma<128, 4><<<GG, 256, 0, stream>>>(Yh, wpk[9], low_enh, nullptr, 0);

  // ---- high path: bneck hr on (high + guide) ----
  add2<<<GE, 256, 0, stream>>>(high, guide, hg, hg_h);
  gemm_wmma<64, 8><<<GG, 256, 0, stream>>>(hg_h, wpk[4], T, nullptr, 0);  // [N,128]
  groupnorm(T, 128, FIN(16), FIN(17), Th_a);
  dw3x3_k<128><<<GE2, 256, 0, stream>>>(Th_a, FIN(18), CAg);
  groupnorm(CAg, 128, FIN(19), FIN(20), Th_b);
  gemm_wmma<128, 4><<<GG, 256, 0, stream>>>(Th_b, wpk[5], G64, nullptr, 1);
  add2<<<GE, 256, 0, stream>>>(G64, hg, high_enh, nullptr);

  // ---- fuse + channel attention + out ----
  concat_h<<<GE2, 256, 0, stream>>>(low_enh, high_enh, cat_hb);
  gemm_wmma<128, 8><<<GG, 256, 0, stream>>>(cat_hb, wpk[10], FUSE, fuse_h, 0);
  gemm_wmma<128, 8><<<GG, 256, 0, stream>>>(fuse_h, wpk[11], CAg, ca_h, 1);
  dw3x3_k<128><<<GE2, 256, 0, stream>>>(ca_h, FIN(36), CAdw);
  ca_apply<<<GE2, 256, 0, stream>>>(CAdw, FUSE, f2_h);
  gemm_wmma<128, 8><<<GG, 256, 0, stream>>>(f2_h, wpk[12], OUTG, nullptr, 0);
  final_add<<<GE2, 256, 0, stream>>>(x, OUTG, (float*)d_out);
#undef FIN
}
